// CustomGRUCell_62027917689184
// MI455X (gfx1250) — compile-verified, matrix operand order checked by reading, not run on hardware
//
#include <hip/hip_runtime.h>

typedef __attribute__((ext_vector_type(16))) __bf16 v16bf;
typedef __attribute__((ext_vector_type(8)))  __bf16 v8bf;
typedef __attribute__((ext_vector_type(4)))  __bf16 v4bf;
typedef __attribute__((ext_vector_type(8)))  float  v8f;
typedef __attribute__((ext_vector_type(4)))  float  v4f;

#define GRU_H   1024
#define GRU_B   8192
#define BM      64
#define BN      64
#define BK      32
#define NK      (GRU_H / BK)  // 32 K-steps
#define LDK     40            // padded LDS row stride in bf16 elements (80 B)
#define SEG     (64 * LDK)    // bf16 elements per 64x32 tile segment
#define BUFSZ   (8 * SEG)     // one pipeline buffer: 8 segments

__device__ __forceinline__ v16bf load_frag(const __bf16* p) {
    v16bf f;
    *(v8bf*)&f       = *(const v8bf*)(p);        // k = koff..koff+7
    *((v8bf*)&f + 1) = *(const v8bf*)(p + 16);   // k = 16+koff..16+koff+7
    return f;
}

__device__ __forceinline__ v4bf cvt4(v4f t) {
    v4bf b4;
    b4.x = (__bf16)t.x; b4.y = (__bf16)t.y;
    b4.z = (__bf16)t.z; b4.w = (__bf16)t.w;
    return b4;
}

__global__ __launch_bounds__(256, 2)
void gru_fused_wmma(const float* __restrict__ x,
                    const float* __restrict__ h,
                    const float* __restrict__ Wih,   // (3H, H)
                    const float* __restrict__ Wb,    // (3H,)
                    const float* __restrict__ Ur,    // (H, H)
                    const float* __restrict__ Uz,    // (H, H)
                    const float* __restrict__ Un,    // (H, H)
                    const float* __restrict__ Unb,   // (H,)
                    float* __restrict__ out)         // (B, H)
{
    // double-buffered: segs per buffer: 0=x 1=h 2=Wr 3=Wz 4=Wn 5=Ur 6=Uz 7=Un
    __shared__ __bf16 lds[2 * BUFSZ];

    const int tid  = threadIdx.x;
    const int lane = tid & 31;
    const int wave = tid >> 5;

    const int bm = blockIdx.x;        // 0..127  (B/64)
    const int bn = blockIdx.y;        // 0..15   (H/64)

    // ---- loader setup: wave w fills segment w (64 rows x 32 cols, f32->bf16)
    const float* src;
    int row0;
    switch (wave) {
        case 0:  src = x;   row0 = bm * BM;            break;
        case 1:  src = h;   row0 = bm * BM;            break;
        case 2:  src = Wih; row0 = 0*GRU_H + bn * BN;  break;
        case 3:  src = Wih; row0 = 1*GRU_H + bn * BN;  break;
        case 4:  src = Wih; row0 = 2*GRU_H + bn * BN;  break;
        case 5:  src = Ur;  row0 = bn * BN;            break;
        case 6:  src = Uz;  row0 = bn * BN;            break;
        default: src = Un;  row0 = bn * BN;            break;
    }
    const int    lrow = lane >> 3;        // 0..3
    const int    lcol = (lane & 7) * 4;   // 0,4,...,28
    const float* gsrc = &src[(size_t)(row0 + lrow) * GRU_H + lcol];

    // ---- compute-side setup: wave grid 2(M) x 4(N); wave tile = 32(M) x 16(N)
    const int wm2  = wave & 1;         // M half within block (2 x 16-row subtiles)
    const int wn4  = wave >> 1;        // N tile within block (16 cols)
    const int lm   = lane & 15;        // row (A) / col (B,C,D) within 16
    const int koff = (lane >> 4) * 8;  // K sub-group select per ISA layout

    const int segStoreOff = wave * SEG + lrow * LDK + lcol;
    const int aX0 = 0 * SEG + (wm2 * 32 +  0 + lm) * LDK + koff;
    const int aX1 = 0 * SEG + (wm2 * 32 + 16 + lm) * LDK + koff;
    const int aH0 = 1 * SEG + (wm2 * 32 +  0 + lm) * LDK + koff;
    const int aH1 = 1 * SEG + (wm2 * 32 + 16 + lm) * LDK + koff;
    const int bOff0 = (wn4 * 16 + lm) * LDK + koff;   // + (2+g)*SEG

    const v8f zacc = {0.f,0.f,0.f,0.f,0.f,0.f,0.f,0.f};
    v8f acc[6][2];                     // [gate][m-subtile]
#pragma unroll
    for (int g = 0; g < 6; ++g)
#pragma unroll
        for (int s = 0; s < 2; ++s)
            acc[g][s] = zacc;

    v4bf st[16];                       // bf16 staging: 32 VGPRs

    // ---------- prologue: load K-tile 0, convert, store to buffer 0 ----------
#pragma unroll
    for (int p = 0; p < 16; ++p)
        st[p] = cvt4(*(const v4f*)&gsrc[(size_t)(p * 4) * GRU_H]);
#pragma unroll
    for (int p = 0; p < 16; ++p)
        *(v4bf*)&lds[segStoreOff + p * 4 * LDK] = st[p];   // ds_store_b64
    __syncthreads();

    for (int kt = 0; kt < NK; ++kt) {
        const int  cur = (kt & 1) * BUFSZ;
        const int  nxt = ((kt + 1) & 1) * BUFSZ;
        const bool pf  = (kt + 1) < NK;

        // ---- issue next K-tile's global loads first (latency under compute);
        //      cvt to bf16 right away to release load registers
        if (pf) {
            const float* gp = gsrc + (size_t)(kt + 1) * BK;
#pragma unroll
            for (int p = 0; p < 16; ++p)
                st[p] = cvt4(*(const v4f*)&gp[(size_t)(p * 4) * GRU_H]);
        }

        // ---- compute phase: 12 WMMAs; each B fragment feeds 2 WMMAs ----
        {
            const __bf16* base = &lds[cur];
            const v16bf ax0 = load_frag(base + aX0);
            const v16bf ax1 = load_frag(base + aX1);
            const v16bf ah0 = load_frag(base + aH0);
            const v16bf ah1 = load_frag(base + aH1);

            v16bf bcur = load_frag(base + 2 * SEG + bOff0);   // gate 0
            v16bf bnxt;
#pragma unroll
            for (int g = 0; g < 6; ++g) {
                if (g < 5)
                    bnxt = load_frag(base + (3 + g) * SEG + bOff0);
                const v16bf a0 = (g < 3) ? ax0 : ah0;
                const v16bf a1 = (g < 3) ? ax1 : ah1;
                acc[g][0] = __builtin_amdgcn_wmma_f32_16x16x32_bf16(
                    false, a0, false, bcur, (short)0, acc[g][0], false, false);
                acc[g][1] = __builtin_amdgcn_wmma_f32_16x16x32_bf16(
                    false, a1, false, bcur, (short)0, acc[g][1], false, false);
                bcur = bnxt;
            }
        }

        // ---- store next tile into the other buffer, single barrier ----
        if (pf) {
#pragma unroll
            for (int p = 0; p < 16; ++p)
                *(v4bf*)&lds[nxt + segStoreOff + p * 4 * LDK] = st[p];
            __syncthreads();
        }
    }

    // ---------- fused GRU epilogue ----------
    {
        const int n   = bn * BN + wn4 * 16 + lm;   // output column (fixed per lane)
        const float brx = Wb[n];
        const float bzx = Wb[GRU_H + n];
        const float bnx = Wb[2 * GRU_H + n];
        const float bnh = Unb[n];
#pragma unroll
        for (int s = 0; s < 2; ++s) {
#pragma unroll
            for (int v = 0; v < 8; ++v) {
                const int m = bm * BM + wm2 * 32 + s * 16 + v + koff;  // output row
                const float rx = acc[0][s][v] + brx;
                const float zx = acc[1][s][v] + bzx;
                const float nx = acc[2][s][v] + bnx;
                const float rh = acc[3][s][v];
                const float zh = acc[4][s][v];
                const float nh = acc[5][s][v] + bnh;
                const float r  = 1.f / (1.f + __expf(-(rx + rh)));
                const float z  = 1.f / (1.f + __expf(-(zx + zh)));
                const float t  = nx + r * nh;
                const float nn = 2.f / (1.f + __expf(-2.f * t)) - 1.f;  // tanh(t)
                const float hv = h[(size_t)m * GRU_H + n];
                out[(size_t)m * GRU_H + n] = (1.f - z) * nn + z * hv;
            }
        }
    }
}

extern "C" void kernel_launch(void* const* d_in, const int* in_sizes, int n_in,
                              void* d_out, int out_size, void* d_ws, size_t ws_size,
                              hipStream_t stream)
{
    (void)in_sizes; (void)n_in; (void)out_size; (void)d_ws; (void)ws_size;
    const float* x   = (const float*)d_in[0];
    const float* h   = (const float*)d_in[1];
    const float* Wih = (const float*)d_in[2];
    const float* Wb  = (const float*)d_in[3];
    const float* Ur  = (const float*)d_in[4];
    const float* Uz  = (const float*)d_in[5];
    const float* Un  = (const float*)d_in[6];
    const float* Unb = (const float*)d_in[7];

    dim3 grid(GRU_B / BM, GRU_H / BN);   // 128 x 16
    gru_fused_wmma<<<grid, 256, 0, stream>>>(x, h, Wih, Wb, Ur, Uz, Un, Unb,
                                             (float*)d_out);
}
